// NlosPoseSformer_549755814409
// MI455X (gfx1250) — matrix operand order repaired and checked by reading, not compiled, not run on hardware
//
#include <hip/hip_runtime.h>

// ---------------- types ----------------
typedef __bf16 bf16;
typedef __attribute__((ext_vector_type(16))) __bf16 v16bf;
typedef __attribute__((ext_vector_type(8)))  float  v8f;
typedef __attribute__((ext_vector_type(4)))  unsigned int v4u;
typedef __attribute__((ext_vector_type(8)))  int v8i;
typedef __attribute__((ext_vector_type(4)))  int v4i;

union AFrag { v16bf v; v4u q[2]; };

#if defined(__has_builtin)
#if __has_builtin(__builtin_amdgcn_tensor_load_to_lds) && __has_builtin(__builtin_amdgcn_s_wait_tensorcnt)
#define USE_TDM 1
#endif
#endif

// ---------------- model constants ----------------
constexpr int B_    = 2;
constexpr int F_    = 16;
constexpr int WPATCH= 14;
constexpr int NPF   = 196;  // patches per frame
constexpr int NJ    = 24;
constexpr int TOK   = NJ + F_ * NPF;       // 3160 tokens per batch
constexpr int NTOT  = B_ * TOK;            // 6320 rows
constexpr int NPATCH= B_ * F_ * NPF;       // 6272 patch rows
constexpr int DIM   = 512;
constexpr int HEADS = 8;
constexpr int DH    = 64;
constexpr int INNER = HEADS * DH;          // 512
constexpr int QKVD  = 3 * INNER;           // 1536
constexpr int FFD   = DIM * 8;             // 4096
constexpr int FFH   = DIM * 4;             // 2048
constexpr int DEPTH = 12;
constexpr int ODIM  = 384;

// ---------------- helpers ----------------
__device__ __forceinline__ bf16 f2bf(float f) {
  union { float f; unsigned u; } x; x.f = f;
  unsigned r = x.u + 0x7fffu + ((x.u >> 16) & 1u);
  unsigned short h = (unsigned short)(r >> 16);
  return __builtin_bit_cast(bf16, h);
}
__device__ __forceinline__ float bf2f(bf16 v) {
  unsigned short h = __builtin_bit_cast(unsigned short, v);
  union { unsigned u; float f; } x; x.u = ((unsigned)h) << 16;
  return x.f;
}
__device__ __forceinline__ float wave_sum(float v) {
#pragma unroll
  for (int m = 16; m >= 1; m >>= 1) v += __shfl_xor(v, m, 32);
  return v;
}
// axial rotary sin/cos for pair index p (0..31), position pos (0..195)
__device__ __forceinline__ void rope_sc(int p, int pos, float& s, float& c) {
  const float L2_5 = 2.3219280948873623f;       // log2(MAX_FREQ/2) = log2(5)
  int i = (p < 16) ? p : (p - 16);
  float scale = exp2f((float)i * (L2_5 / 15.f));
  int hp = pos / WPATCH, wp = pos - hp * WPATCH;
  float lin = ((p < 16) ? (float)hp : (float)wp) * (2.f / 13.f) - 1.f;
  float th = lin * scale * 3.14159265358979323f;
  __sincosf(th, &s, &c);
}

#if USE_TDM
// Issue one 2D TDM tile load: global (row-major, stride0 elems) -> LDS,
// bf16 elements, rows of `tile0` elems padded to tile0+8 elems in LDS
// (pad_interval = 16 DWORDs = 32 bf16 = tile row; pad_amount = 4 DWORDs = 8 bf16).
// dim0_rem/dim1_rem: remaining tensor extent from the tile start (OOB -> zero fill).
__device__ __forceinline__ void tdm_load_2d(const void* gaddr, unsigned lds_off,
                                            int dim0_rem, int dim1_rem,
                                            long long stride0, int tile0, int tile1) {
  unsigned long long ga = (unsigned long long)gaddr;
  v4u g0;
  g0.x = 1u;                                            // count=1, user descriptor
  g0.y = lds_off;                                       // LDS byte address
  g0.z = (unsigned)(ga & 0xffffffffu);                  // global_addr[31:0]
  g0.w = (unsigned)((ga >> 32) & 0x1ffffffu) | (2u << 30); // addr[56:32] | type=2
  unsigned d0 = (unsigned)dim0_rem, d1 = (unsigned)dim1_rem;
  unsigned long long s0 = (unsigned long long)stride0;
  v8i g1;
  g1[0] = (int)((1u << 16)      // data_size = 2 bytes
              | (1u << 20)      // pad_enable
              | (3u << 22)      // pad_interval: 16 DWORDs
              | (3u << 25));    // pad_amount: 4 DWORDs
  g1[1] = (int)((d0 & 0xffffu) << 16);                        // dim0[15:0] @ [63:48]
  g1[2] = (int)((d0 >> 16) | ((d1 & 0xffffu) << 16));         // dim0[31:16], dim1[15:0]
  g1[3] = (int)((d1 >> 16) | ((unsigned)tile0 << 16));        // dim1[31:16], tile_dim0
  g1[4] = (int)(unsigned)tile1;                               // tile_dim1 (tile_dim2=0)
  g1[5] = (int)(unsigned)(s0 & 0xffffffffu);                  // stride0[31:0]
  g1[6] = (int)(unsigned)((s0 >> 32) & 0xffffu);              // stride0[47:32]
  g1[7] = 0;
  v4i z4 = {0, 0, 0, 0};
  v8i z8 = {0, 0, 0, 0, 0, 0, 0, 0};
  // clang-23 / therock form: (g0, g1, g2, g3, g4, cpol)
  __builtin_amdgcn_tensor_load_to_lds(g0, g1, z4, z4, z8, 0);
}
#endif

// ---------------- elementwise kernels ----------------
// tiled transpose + convert: in [K][N] fp32 (slice z) -> out [N][K] bf16
__global__ __launch_bounds__(256) void k_cvt_t(const float* __restrict__ in,
                                               bf16* __restrict__ out, int K, int N) {
  __shared__ float tile[32][33];
  const float* src = in + (long long)blockIdx.z * K * N;
  bf16* dst = out + (long long)blockIdx.z * K * N;
  int n0 = blockIdx.x * 32, k0 = blockIdx.y * 32;
  int tx = threadIdx.x & 31, ty = threadIdx.x >> 5;
#pragma unroll
  for (int r = 0; r < 4; ++r)
    tile[ty + r * 8][tx] = src[(long long)(k0 + ty + r * 8) * N + n0 + tx];
  __syncthreads();
#pragma unroll
  for (int r = 0; r < 4; ++r)
    dst[(long long)(n0 + ty + r * 8) * K + k0 + tx] = f2bf(tile[tx][ty + r * 8]);
}

// video (2,16,2,224,224) -> A (6272, 512) bf16, feature = (py*16+px)*2 + ch
__global__ __launch_bounds__(256) void k_patch_gather(const float* __restrict__ video,
                                                      bf16* __restrict__ A) {
  long long i = (long long)blockIdx.x * 256 + threadIdx.x;   // < 6272*512
  int d = (int)(i & 511); long long r = i >> 9;
  int t = (int)(r % 3136); int bi = (int)(r / 3136);
  int fi = t / NPF; int pp = t - fi * NPF;
  int hpi = pp / WPATCH, wpi = pp - hpi * WPATCH;
  int ch = d & 1; int pix = d >> 1;
  int py = pix >> 4, px = pix & 15;
  long long off = ((((long long)bi * F_ + fi) * 2 + ch) * 224 + (hpi * 16 + py)) * 224
                + (wpi * 16 + px);
  A[i] = f2bf(video[off]);
}

// build x (6320,512): first 24 rows per batch = joints_token, rest = tokens_tmp
__global__ __launch_bounds__(256) void k_build_x(const float* __restrict__ joints,
                                                 const float* __restrict__ tok,
                                                 float* __restrict__ x) {
  long long i = (long long)blockIdx.x * 256 + threadIdx.x;   // < 6320*512
  int c = (int)(i & 511); long long r = i >> 9;
  int ri = (int)(r % TOK); int bi = (int)(r / TOK);
  x[i] = (ri < NJ) ? joints[ri * DIM + c]
                   : tok[((long long)bi * 3136 + (ri - NJ)) * DIM + c];
}

// LayerNorm over 512 cols, wave per row, fp32 in -> bf16 out
__global__ __launch_bounds__(256) void k_ln(const float* __restrict__ x,
                                            const float* __restrict__ w,
                                            const float* __restrict__ b,
                                            bf16* __restrict__ out, int rows) {
  int row = blockIdx.x * 8 + (threadIdx.x >> 5);
  if (row >= rows) return;
  int lane = threadIdx.x & 31;
  const float* xr = x + (long long)row * DIM;
  float v[16], s = 0.f, s2 = 0.f;
#pragma unroll
  for (int t = 0; t < 16; ++t) { v[t] = xr[lane + t * 32]; s += v[t]; s2 += v[t] * v[t]; }
  s = wave_sum(s); s2 = wave_sum(s2);
  float mean = s * (1.f / DIM);
  float rstd = rsqrtf(s2 * (1.f / DIM) - mean * mean + 1e-5f);
  bf16* orow = out + (long long)row * DIM;
#pragma unroll
  for (int t = 0; t < 16; ++t) {
    int c = lane + t * 32;
    orow[c] = f2bf((v[t] - mean) * rstd * w[c] + b[c]);
  }
}

// LN over the 48 joint-token rows (gathered), contiguous bf16 out (48,512)
__global__ __launch_bounds__(256) void k_ln_joints(const float* __restrict__ x,
                                                   const float* __restrict__ w,
                                                   const float* __restrict__ b,
                                                   bf16* __restrict__ out) {
  int row = blockIdx.x * 8 + (threadIdx.x >> 5);      // 0..47
  if (row >= B_ * NJ) return;
  int lane = threadIdx.x & 31;
  int bi = row / NJ, j = row - bi * NJ;
  const float* xr = x + ((long long)bi * TOK + j) * DIM;
  float v[16], s = 0.f, s2 = 0.f;
#pragma unroll
  for (int t = 0; t < 16; ++t) { v[t] = xr[lane + t * 32]; s += v[t]; s2 += v[t] * v[t]; }
  s = wave_sum(s); s2 = wave_sum(s2);
  float mean = s * (1.f / DIM);
  float rstd = rsqrtf(s2 * (1.f / DIM) - mean * mean + 1e-5f);
  bf16* orow = out + (long long)row * DIM;
#pragma unroll
  for (int t = 0; t < 16; ++t) {
    int c = lane + t * 32;
    orow[c] = f2bf((v[t] - mean) * rstd * w[c] + b[c]);
  }
}

// gated GELU: h1 (6320,4096) bf16 -> g (6320,2048) bf16 : a * gelu_exact(g)
__global__ __launch_bounds__(256) void k_gate(const bf16* __restrict__ h1,
                                              bf16* __restrict__ out) {
  long long i = (long long)blockIdx.x * 256 + threadIdx.x;   // < 6320*2048
  int c = (int)(i & (FFH - 1)); long long r = i >> 11;
  float a = bf2f(h1[r * FFD + c]);
  float g = bf2f(h1[r * FFD + FFH + c]);
  float ge = 0.5f * g * (1.f + erff(g * 0.7071067811865475f));
  out[i] = f2bf(a * ge);
}

// ---------------- WMMA GEMM ----------------
// C[M,N] = A[M,K] @ Bt[N,K]^T (+bias) (+residual), A/Bt bf16, accum fp32.
// Block: 256 thr = 8 waves; tile 64x128; wave tile 32x32 via 4x wmma 16x16x32 bf16.
// Tiles staged into dynamic LDS (double-buffered) by the Tensor Data Mover when
// available (TENSORcnt pipelined), else by vectorized manual staging.
constexpr int GBM = 64, GBN = 128, GBK = 32, GLDT = 40;
constexpr unsigned GSZA = GBM * GLDT * 2;   // bytes per A buffer
constexpr unsigned GSZB = GBN * GLDT * 2;   // bytes per B buffer
constexpr unsigned GSMEM = 2 * (GSZA + GSZB);

template<bool HAS_BIAS, bool RESID, bool OUT_BF16>
__global__ __launch_bounds__(256, 2) void gemm_bf16_kernel(
    const bf16* __restrict__ A, const bf16* __restrict__ Bt,
    const float* __restrict__ bias, const float* __restrict__ resid,
    void* __restrict__ Cout, int M, int N, int K)
{
  extern __shared__ char smem[];               // dynamic-only => segment offset 0
  bf16* lA = (bf16*)smem;                      // 2 x (64 x 40)
  bf16* lB = (bf16*)(smem + 2 * GSZA);         // 2 x (128 x 40), [n][k]
  const int tid  = threadIdx.x;
  const int wid  = tid >> 5, lane = tid & 31;
  const int lh   = lane >> 4, lm = lane & 15;
  const int m0   = blockIdx.y * GBM, n0 = blockIdx.x * GBN;
  const int wm   = (wid & 1) * 32, wn = (wid >> 1) * 32;

  v8f acc[2][2] = {};

  // fragments per ISA layout:
  // A: lane holds row m=lane%16; half h: two 8-elem chunks at k = h*8, 16+h*8
  // B: lane holds col n=lane%16; half h: two 8-elem chunks at k = h*8, 16+h*8
  auto mma_step = [&](const bf16* bufA, const bf16* bufB) {
    AFrag a0, a1, b0, b1;
    const bf16* pa = bufA + (wm + lm) * GLDT;
    a0.q[0] = *(const v4u*)(pa + lh * 8);
    a0.q[1] = *(const v4u*)(pa + 16 + lh * 8);
    const bf16* pa1 = pa + 16 * GLDT;
    a1.q[0] = *(const v4u*)(pa1 + lh * 8);
    a1.q[1] = *(const v4u*)(pa1 + 16 + lh * 8);
    const bf16* pb = bufB + (wn + lm) * GLDT;
    b0.q[0] = *(const v4u*)(pb + lh * 8);
    b0.q[1] = *(const v4u*)(pb + 16 + lh * 8);
    const bf16* pb1 = pb + 16 * GLDT;
    b1.q[0] = *(const v4u*)(pb1 + lh * 8);
    b1.q[1] = *(const v4u*)(pb1 + 16 + lh * 8);
    acc[0][0] = __builtin_amdgcn_wmma_f32_16x16x32_bf16(false, a0.v, false, b0.v, (short)0, acc[0][0], false, false);
    acc[0][1] = __builtin_amdgcn_wmma_f32_16x16x32_bf16(false, a0.v, false, b1.v, (short)0, acc[0][1], false, false);
    acc[1][0] = __builtin_amdgcn_wmma_f32_16x16x32_bf16(false, a1.v, false, b0.v, (short)0, acc[1][0], false, false);
    acc[1][1] = __builtin_amdgcn_wmma_f32_16x16x32_bf16(false, a1.v, false, b1.v, (short)0, acc[1][1], false, false);
  };

#if USE_TDM
  const int steps = K / GBK;
  if (wid == 0) {   // TDM: EXEC-independent, issue from wave 0 only
    tdm_load_2d(A  + (long long)m0 * K, 0u,          K, M - m0, K, GBK, GBM);
    tdm_load_2d(Bt + (long long)n0 * K, 2u * GSZA,   K, N - n0, K, GBK, GBN);
  }
  for (int s = 0; s < steps; ++s) {
    const int cur = s & 1;
    if (wid == 0) {
      if (s + 1 < steps) {   // overlap next-tile DMA with this tile's WMMA
        const int k1 = (s + 1) * GBK;
        const unsigned nb = (unsigned)(1 - cur);
        tdm_load_2d(A  + (long long)m0 * K + k1, nb * GSZA,             K - k1, M - m0, K, GBK, GBM);
        tdm_load_2d(Bt + (long long)n0 * K + k1, 2u * GSZA + nb * GSZB, K - k1, N - n0, K, GBK, GBN);
        __builtin_amdgcn_s_wait_tensorcnt(2);   // oldest pair (cur) complete
      } else {
        __builtin_amdgcn_s_wait_tensorcnt(0);
      }
    }
    __syncthreads();
    mma_step(lA + cur * (GSZA / 2), lB + cur * (GSZB / 2));
    __syncthreads();
  }
#else
  const int arow = tid >> 2, akc = (tid & 3) * 8;        // A tile: 64 x 32
  for (int k0 = 0; k0 < K; k0 += GBK) {
    {
      int gm = m0 + arow; if (gm >= M) gm = M - 1;
      v4u d = *(const v4u*)(A + (long long)gm * K + k0 + akc);
      *(v4u*)(lA + arow * GLDT + akc) = d;
      if (k0 + GBK < K)
        __builtin_prefetch(A + (long long)gm * K + k0 + GBK + akc, 0, 1);
    }
#pragma unroll
    for (int r = 0; r < 2; ++r) {
      int idx = tid * 2 + r;                             // 0..511
      int nr = idx >> 2, kc = (idx & 3) * 8;
      int gn = n0 + nr; if (gn >= N) gn = N - 1;
      v4u d = *(const v4u*)(Bt + (long long)gn * K + k0 + kc);
      *(v4u*)(lB + nr * GLDT + kc) = d;
    }
    __syncthreads();
    mma_step(lA, lB);
    __syncthreads();
  }
#endif

  // epilogue: element r of acc is C[m0+wm+mi*16 + r + lh*8][n0+wn+ni*16+lm]
  const bool full = (m0 + GBM <= M);
#pragma unroll
  for (int mi = 0; mi < 2; ++mi) {
#pragma unroll
    for (int ni = 0; ni < 2; ++ni) {
      const int n = n0 + wn + ni * 16 + lm;
      const float bv = HAS_BIAS ? bias[n] : 0.f;
      const int mbase = m0 + wm + mi * 16 + lh * 8;
      if (full) {
#pragma unroll
        for (int r = 0; r < 8; ++r) {
          long long off = (long long)(mbase + r) * N + n;
          float v = acc[mi][ni][r] + bv;
          if (RESID) v += resid[off];
          if (OUT_BF16) ((bf16*)Cout)[off] = f2bf(v);
          else          ((float*)Cout)[off] = v;
        }
      } else {
#pragma unroll
        for (int r = 0; r < 8; ++r) {
          if (mbase + r < M) {
            long long off = (long long)(mbase + r) * N + n;
            float v = acc[mi][ni][r] + bv;
            if (RESID) v += resid[off];
            if (OUT_BF16) ((bf16*)Cout)[off] = f2bf(v);
            else          ((float*)Cout)[off] = v;
          }
        }
      }
    }
  }
}

// ---------------- attention (wave-per-query online softmax) ----------------
__global__ __launch_bounds__(256) void k_attn_joints(const float* __restrict__ qkv,
                                                     bf16* __restrict__ outb) {
  int w = blockIdx.x * 8 + (threadIdx.x >> 5);          // 0..383
  int lane = threadIdx.x & 31;
  int jq = w % NJ; int t = w / NJ;
  int h = t & 7; int bi = t >> 3;
  const float* base = qkv + (long long)bi * TOK * QKVD;
  int d0 = lane * 2;
  const float* qp = base + (long long)jq * QKVD + h * DH + d0;
  float q0 = qp[0] * 0.125f, q1 = qp[1] * 0.125f;
  float m = -1e30f, l = 0.f, a0 = 0.f, a1 = 0.f;
  for (int tok = 0; tok < TOK; ++tok) {
    const float* kp = base + (long long)tok * QKVD + INNER + h * DH + d0;
    float sc = wave_sum(q0 * kp[0] + q1 * kp[1]);
    float mn = fmaxf(m, sc);
    float corr = __expf(m - mn), p = __expf(sc - mn);
    const float* vp = base + (long long)tok * QKVD + 2 * INNER + h * DH + d0;
    l = l * corr + p;
    a0 = a0 * corr + p * vp[0];
    a1 = a1 * corr + p * vp[1];
    m = mn;
  }
  float inv = 1.f / l;
  bf16* op = outb + ((long long)bi * TOK + jq) * DIM + h * DH + d0;
  op[0] = f2bf(a0 * inv); op[1] = f2bf(a1 * inv);
}

__global__ __launch_bounds__(256) void k_attn_spatial(const float* __restrict__ qkv,
                                                      bf16* __restrict__ outb) {
  int w = blockIdx.x * 8 + (threadIdx.x >> 5);          // < 50176
  int lane = threadIdx.x & 31;
  int qpos = w % NPF; int t = w / NPF;
  int fi = t & 15; t >>= 4;
  int h = t & 7; int bi = t >> 3;
  const float* base = qkv + (long long)bi * TOK * QKVD;
  int qtok = NJ + fi * NPF + qpos;
  int d0 = lane * 2;
  const float* qp = base + (long long)qtok * QKVD + h * DH + d0;
  float qe = qp[0] * 0.125f, qo = qp[1] * 0.125f;
  float s, c; rope_sc(lane, qpos, s, c);
  float q0 = qe * c - qo * s, q1 = qo * c + qe * s;
  float m = -1e30f, l = 0.f, a0 = 0.f, a1 = 0.f;
  for (int j = 0; j < NJ + NPF; ++j) {
    int tok = (j < NJ) ? j : (NJ + fi * NPF + (j - NJ));
    const float* kp = base + (long long)tok * QKVD + INNER + h * DH + d0;
    float ke = kp[0], ko = kp[1];
    if (j >= NJ) {
      float ks, kc; rope_sc(lane, j - NJ, ks, kc);
      float t0 = ke * kc - ko * ks;
      ko = ko * kc + ke * ks; ke = t0;
    }
    float sc = wave_sum(q0 * ke + q1 * ko);
    float mn = fmaxf(m, sc);
    float corr = __expf(m - mn), p = __expf(sc - mn);
    const float* vp = base + (long long)tok * QKVD + 2 * INNER + h * DH + d0;
    l = l * corr + p;
    a0 = a0 * corr + p * vp[0];
    a1 = a1 * corr + p * vp[1];
    m = mn;
  }
  float inv = 1.f / l;
  bf16* op = outb + ((long long)bi * TOK + qtok) * DIM + h * DH + d0;
  op[0] = f2bf(a0 * inv); op[1] = f2bf(a1 * inv);
}

// ---------------- host ----------------
static inline size_t alignup(size_t v) { return (v + 255) & ~(size_t)255; }

extern "C" void kernel_launch(void* const* d_in, const int* in_sizes, int n_in,
                              void* d_out, int out_size, void* d_ws, size_t ws_size,
                              hipStream_t stream) {
  const float* video        = (const float*)d_in[0];
  const float* patch_w      = (const float*)d_in[1];
  const float* patch_b      = (const float*)d_in[2];
  const float* joints_token = (const float*)d_in[3];
  const float* ln_attn_w    = (const float*)d_in[4];
  const float* ln_attn_b    = (const float*)d_in[5];
  const float* qkv_w        = (const float*)d_in[6];
  const float* attn_out_w   = (const float*)d_in[7];
  const float* attn_out_b   = (const float*)d_in[8];
  const float* ln_ff_w      = (const float*)d_in[9];
  const float* ln_ff_b      = (const float*)d_in[10];
  const float* ff1_w        = (const float*)d_in[11];
  const float* ff1_b        = (const float*)d_in[12];
  const float* ff2_w        = (const float*)d_in[13];
  const float* ff2_b        = (const float*)d_in[14];
  const float* ln_out_w     = (const float*)d_in[15];
  const float* ln_out_b     = (const float*)d_in[16];
  const float* out_w        = (const float*)d_in[17];
  const float* out_b        = (const float*)d_in[18];
  float* dout = (float*)d_out;

  // workspace carve-up (regions time-sliced; ~183 MiB total)
  char* ws = (char*)d_ws;
  size_t o = 0;
  auto carve = [&](size_t bytes) { char* p = ws + o; o += alignup(bytes); return p; };
  bf16*  w_patch = (bf16*)carve((size_t)DIM * DIM * 2);        // transposed [N][K]
  bf16*  w_head  = (bf16*)carve((size_t)DIM * ODIM * 2);
  bf16*  w_qkv   = (bf16*)carve((size_t)DEPTH * DIM * QKVD * 2);
  bf16*  w_out   = (bf16*)carve((size_t)DEPTH * INNER * DIM * 2);
  bf16*  w_ff1   = (bf16*)carve((size_t)DEPTH * DIM * FFD * 2);
  bf16*  w_ff2   = (bf16*)carve((size_t)DEPTH * FFH * DIM * 2);
  float* x       = (float*)carve((size_t)NTOT * DIM * 4);
  char*  big     = carve((size_t)NTOT * FFD * 2 > (size_t)NTOT * QKVD * 4
                         ? (size_t)NTOT * FFD * 2 : (size_t)NTOT * QKVD * 4);
  char*  act     = carve((size_t)NTOT * FFH * 2);
  if (ws_size < o) return;

  float* tokens_tmp = (float*)big;   // time-sliced: tokens fp32 / qkv fp32 / h1 bf16
  float* qkvbuf     = (float*)big;
  bf16*  h1         = (bf16*)big;
  bf16*  Abuf       = (bf16*)act;    // time-sliced: patches / xln / attn-out / gate
  bf16*  xln        = (bf16*)act;
  bf16*  attn_out   = (bf16*)act;
  bf16*  gbuf       = (bf16*)act;
  bf16*  jln        = (bf16*)act;

  // weight transpose+convert (fp32 [K][N] -> bf16 [N][K]); z = layer
  auto cvt_t = [&](const float* src, bf16* dst, int K, int N, int layers) {
    k_cvt_t<<<dim3(N / 32, K / 32, layers), dim3(256), 0, stream>>>(src, dst, K, N);
  };
  cvt_t(patch_w,    w_patch, DIM, DIM, 1);
  cvt_t(out_w,      w_head,  DIM, ODIM, 1);
  cvt_t(qkv_w,      w_qkv,   DIM, QKVD, DEPTH);
  cvt_t(attn_out_w, w_out,   INNER, DIM, DEPTH);
  cvt_t(ff1_w,      w_ff1,   DIM, FFD, DEPTH);
  cvt_t(ff2_w,      w_ff2,   FFH, DIM, DEPTH);

  // patch embed
  k_patch_gather<<<dim3(NPATCH * DIM / 256), dim3(256), 0, stream>>>(video, Abuf);
  gemm_bf16_kernel<true, false, false>
      <<<dim3(DIM / GBN, (NPATCH + GBM - 1) / GBM), dim3(256), GSMEM, stream>>>(
          Abuf, w_patch, patch_b, nullptr, tokens_tmp, NPATCH, DIM, DIM);
  k_build_x<<<dim3(NTOT * DIM / 256), dim3(256), 0, stream>>>(joints_token, tokens_tmp, x);

  for (int l = 0; l < DEPTH; ++l) {
    const bf16* wq = w_qkv + (long long)l * DIM * QKVD;
    const bf16* wo = w_out + (long long)l * INNER * DIM;
    const bf16* w1 = w_ff1 + (long long)l * DIM * FFD;
    const bf16* w2 = w_ff2 + (long long)l * FFH * DIM;

    k_ln<<<dim3((NTOT + 7) / 8), dim3(256), 0, stream>>>(
        x, ln_attn_w + l * DIM, ln_attn_b + l * DIM, xln, NTOT);
    gemm_bf16_kernel<false, false, false>
        <<<dim3(QKVD / GBN, (NTOT + GBM - 1) / GBM), dim3(256), GSMEM, stream>>>(
            xln, wq, nullptr, nullptr, qkvbuf, NTOT, QKVD, DIM);

    k_attn_joints <<<dim3(B_ * HEADS * NJ / 8), dim3(256), 0, stream>>>(qkvbuf, attn_out);
    k_attn_spatial<<<dim3(B_ * HEADS * F_ * NPF / 8), dim3(256), 0, stream>>>(qkvbuf, attn_out);

    gemm_bf16_kernel<true, true, false>
        <<<dim3(DIM / GBN, (NTOT + GBM - 1) / GBM), dim3(256), GSMEM, stream>>>(
            attn_out, wo, attn_out_b + l * DIM, x, x, NTOT, DIM, INNER);

    k_ln<<<dim3((NTOT + 7) / 8), dim3(256), 0, stream>>>(
        x, ln_ff_w + l * DIM, ln_ff_b + l * DIM, xln, NTOT);
    gemm_bf16_kernel<true, false, true>
        <<<dim3(FFD / GBN, (NTOT + GBM - 1) / GBM), dim3(256), GSMEM, stream>>>(
            xln, w1, ff1_b + l * FFD, nullptr, h1, NTOT, FFD, DIM);
    k_gate<<<dim3(NTOT * FFH / 256), dim3(256), 0, stream>>>(h1, gbuf);
    gemm_bf16_kernel<true, true, false>
        <<<dim3(DIM / GBN, (NTOT + GBM - 1) / GBM), dim3(256), GSMEM, stream>>>(
            gbuf, w2, ff2_b + l * DIM, x, x, NTOT, DIM, FFH);
  }

  k_ln_joints<<<dim3(6), dim3(256), 0, stream>>>(x, ln_out_w, ln_out_b, jln);
  gemm_bf16_kernel<true, false, false>
      <<<dim3(ODIM / GBN, (B_ * NJ + GBM - 1) / GBM), dim3(256), GSMEM, stream>>>(
          jln, w_head, out_b, nullptr, dout, B_ * NJ, ODIM, DIM);

  (void)in_sizes; (void)n_in; (void)out_size;
}
